// FruitletAssociator_90847148245219
// MI455X (gfx1250) — matrix operand order repaired and checked by reading, not compiled, not run on hardware
//
#include <hip/hip_runtime.h>

typedef unsigned short u16;
typedef __bf16 bf16x16 __attribute__((ext_vector_type(16)));
typedef float  floatx8 __attribute__((ext_vector_type(8)));

#define BATCH   4
#define DIM     512
#define NPTS    1024
#define NHEADS  8
#define HDIM    64
#define NLAYERS 12
#define TWOD    1024
#define MP1     (NPTS + 1)

static __device__ __forceinline__ u16 f2bf(float f) {
  unsigned u = __float_as_uint(f);
  unsigned r = u + 0x7FFFu + ((u >> 16) & 1u);   // round-to-nearest-even
  return (u16)(r >> 16);
}

// Direct global->LDS async copy of 16 bytes (CDNA5 GLOBAL_LOAD_ASYNC_TO_LDS_B128,
// tracked by ASYNCcnt). ldsAddr = low 32 bits of the generic LDS pointer.
static __device__ __forceinline__ void async_g2l_b128(const void* gptr, void* lptr) {
  unsigned ldsAddr = (unsigned)(unsigned long long)lptr;
  unsigned long long ga = (unsigned long long)gptr;
  asm volatile("global_load_async_to_lds_b128 %0, %1, off"
               :: "v"(ldsAddr), "v"(ga) : "memory");
}
static __device__ __forceinline__ void wait_asynccnt0() {
  asm volatile("s_wait_asynccnt 0x0" ::: "memory");
}

union FragBF { uint4 u4[2]; bf16x16 v; };

// ---------------------------------------------------------------------------
// Tiled WMMA GEMM: C[M,N](f32/bf16) = scale * A(bf16) x B(bf16) + bias
//   !AT: A[m*lda+k]   AT: A[k*lda+m]
//   !BT: B[k*ldb+n]   BT: B[n*ldb+k]
//   per-z C offset = (z/zdiv)*sCz1 + (z%zdiv)*sCz2  (head-merge remap support)
// Block: 256 threads = 8 wave32 waves in a 4x2 grid; wave tile 32x64;
// per K-step each wave issues 8x v_wmma_f32_16x16x32_bf16.
// Full tiles (the common case) stage via async global->LDS with no predication.
// ---------------------------------------------------------------------------
constexpr int BM = 128, BN = 128, BK = 32;

template <bool AT, bool BT, bool BIAS, bool WF32, bool WBF>
__global__ __launch_bounds__(256) void gemm_wmma(
    const u16* __restrict__ A, long long sAz, int lda,
    const u16* __restrict__ B, long long sBz, int ldb,
    float* __restrict__ C, u16* __restrict__ Cbf,
    int zdiv, long long sCz1, long long sCz2, int ldc,
    const float* __restrict__ bias, float scale, int M, int N, int K) {
  __shared__ __align__(16) u16 sA[BM * BK];   // [m][k]
  __shared__ __align__(16) u16 sB[BN * BK];   // [n][k]
  const int z = blockIdx.z;
  const u16* Ab = A + (long long)z * sAz;
  const u16* Bb = B + (long long)z * sBz;
  const long long cOff = (long long)(z / zdiv) * sCz1 + (long long)(z % zdiv) * sCz2;
  const int m0 = blockIdx.y * BM, n0 = blockIdx.x * BN;
  const int tid = threadIdx.x;
  const int wave = tid >> 5, lane = tid & 31;
  const int wm = wave >> 1, wn = wave & 1;
  const int laneHalf = lane >> 4, l15 = lane & 15;
  const bool fullM = (m0 + BM <= M);   // block-uniform fast-path predicates
  const bool fullN = (n0 + BN <= N);

  floatx8 acc[2][4] = {};

  for (int k0 = 0; k0 < K; k0 += BK) {
    const bool fullK = (k0 + BK <= K);
    bool usedAsync = false;

    // ---- stage A tile -> sA[m][k] ----
    if (!AT) {
      if (fullM && fullK) {
        usedAsync = true;
        for (int c = tid; c < (BM * BK) / 8; c += 256) {
          int m = c >> 2, kk = (c & 3) * 8;
          async_g2l_b128(Ab + (long long)(m0 + m) * lda + (k0 + kk), &sA[m * BK + kk]);
        }
      } else {
        for (int c = tid; c < (BM * BK) / 8; c += 256) {
          int m = c >> 2, kk = (c & 3) * 8;
          int gm = m0 + m, gk = k0 + kk;
          uint4 val = make_uint4(0u, 0u, 0u, 0u);
          if (gm < M && gk + 7 < K) val = *(const uint4*)(Ab + (long long)gm * lda + gk);
          *(uint4*)(&sA[m * BK + kk]) = val;
        }
      }
    } else {
      if (fullM && fullK) {   // transpose scatter, unpredicated
        for (int c = tid; c < (BM * BK) / 8; c += 256) {
          int k = c >> 4, mm = (c & 15) * 8;
          alignas(16) u16 tmp[8];
          *(uint4*)tmp = *(const uint4*)(Ab + (long long)(k0 + k) * lda + (m0 + mm));
          #pragma unroll
          for (int j = 0; j < 8; ++j) sA[(mm + j) * BK + k] = tmp[j];
        }
      } else {
        for (int c = tid; c < (BM * BK) / 8; c += 256) {
          int k = c >> 4, mm = (c & 15) * 8;
          int gk = k0 + k, gm = m0 + mm;
          alignas(16) u16 tmp[8] = {0, 0, 0, 0, 0, 0, 0, 0};
          if (gk < K) {
            if (gm + 7 < M) *(uint4*)tmp = *(const uint4*)(Ab + (long long)gk * lda + gm);
            else for (int j = 0; j < 8; ++j) if (gm + j < M) tmp[j] = Ab[(long long)gk * lda + gm + j];
          }
          #pragma unroll
          for (int j = 0; j < 8; ++j) sA[(mm + j) * BK + k] = tmp[j];
        }
      }
    }
    // ---- stage B tile -> sB[n][k] ----
    if (BT) {
      if (fullN && fullK) {
        usedAsync = true;
        for (int c = tid; c < (BN * BK) / 8; c += 256) {
          int n = c >> 2, kk = (c & 3) * 8;
          async_g2l_b128(Bb + (long long)(n0 + n) * ldb + (k0 + kk), &sB[n * BK + kk]);
        }
      } else {
        for (int c = tid; c < (BN * BK) / 8; c += 256) {
          int n = c >> 2, kk = (c & 3) * 8;
          int gn = n0 + n, gk = k0 + kk;
          uint4 val = make_uint4(0u, 0u, 0u, 0u);
          if (gn < N && gk + 7 < K) val = *(const uint4*)(Bb + (long long)gn * ldb + gk);
          *(uint4*)(&sB[n * BK + kk]) = val;
        }
      }
    } else {
      if (fullN && fullK) {   // transpose scatter, unpredicated
        for (int c = tid; c < (BN * BK) / 8; c += 256) {
          int k = c >> 4, nn = (c & 15) * 8;
          alignas(16) u16 tmp[8];
          *(uint4*)tmp = *(const uint4*)(Bb + (long long)(k0 + k) * ldb + (n0 + nn));
          #pragma unroll
          for (int j = 0; j < 8; ++j) sB[(nn + j) * BK + k] = tmp[j];
        }
      } else {
        for (int c = tid; c < (BN * BK) / 8; c += 256) {
          int k = c >> 4, nn = (c & 15) * 8;
          int gk = k0 + k, gn = n0 + nn;
          alignas(16) u16 tmp[8] = {0, 0, 0, 0, 0, 0, 0, 0};
          if (gk < K) {
            if (gn + 7 < N) *(uint4*)tmp = *(const uint4*)(Bb + (long long)gk * ldb + gn);
            else for (int j = 0; j < 8; ++j) if (gn + j < N) tmp[j] = Bb[(long long)gk * ldb + gn + j];
          }
          #pragma unroll
          for (int j = 0; j < 8; ++j) sB[(nn + j) * BK + k] = tmp[j];
        }
      }
    }
    // async LDS writes must drain before other waves read past the barrier
    if (usedAsync) wait_asynccnt0();
    __syncthreads();

    // L2 prefetch of next K tile (global_prefetch_b8)
    if (k0 + BK < K && lane == 0) {
      const u16* pa = AT ? (Ab + (long long)(k0 + BK + wave) * lda + m0)
                         : (Ab + (long long)(m0 + wave * 16) * lda + (k0 + BK));
      const u16* pb = BT ? (Bb + (long long)(n0 + wave * 16) * ldb + (k0 + BK))
                         : (Bb + (long long)(k0 + BK + wave) * ldb + n0);
      __builtin_prefetch(pa, 0, 1);
      __builtin_prefetch(pb, 0, 1);
    }

    // ---- fragments per ISA 16-bit layouts ----
    // A 16x32: lanes 0-15 row M=l15 K:{0..7,16..23}; lanes 16-31 K:{8..15,24..31}
    // B 32x16: lane col=l15, lanes 0-15 K:{0..7,16..23}; lanes 16-31 K:{8..15,24..31}
    FragBF a[2], b[4];
    #pragma unroll
    for (int i = 0; i < 2; ++i) {
      const u16* ap = &sA[(wm * 32 + i * 16 + l15) * BK + laneHalf * 8];
      a[i].u4[0] = *(const uint4*)ap;
      a[i].u4[1] = *(const uint4*)(ap + 16);
    }
    #pragma unroll
    for (int j = 0; j < 4; ++j) {
      const u16* bp = &sB[(wn * 64 + j * 16 + l15) * BK + laneHalf * 8];
      b[j].u4[0] = *(const uint4*)bp;
      b[j].u4[1] = *(const uint4*)(bp + 16);
    }
    #pragma unroll
    for (int i = 0; i < 2; ++i)
      #pragma unroll
      for (int j = 0; j < 4; ++j)
        acc[i][j] = __builtin_amdgcn_wmma_f32_16x16x32_bf16(
            false, a[i].v, false, b[j].v, (short)0, acc[i][j], false, false);
    __syncthreads();
  }

  // C 16x16 layout: lane l, vgpr e -> row = base + (l>>4)*8 + e, col = base + (l&15)
  for (int i = 0; i < 2; ++i) {
    for (int j = 0; j < 4; ++j) {
      const int rBase = m0 + wm * 32 + i * 16 + laneHalf * 8;
      const int col = n0 + wn * 64 + j * 16 + l15;
      if (col >= N) continue;
      #pragma unroll
      for (int e = 0; e < 8; ++e) {
        const int row = rBase + e;
        if (row >= M) continue;
        float vv = acc[i][j][e] * scale;
        if constexpr (BIAS) vv += bias[row];
        const long long o = cOff + (long long)row * ldc + col;
        if constexpr (WF32) C[o] = vv;
        if constexpr (WBF) Cbf[o] = f2bf(vv);
      }
    }
  }
}

// ---------------------------------------------------------------------------
// Elementwise / reduction kernels
// ---------------------------------------------------------------------------
__global__ void cast_bf16(const float* __restrict__ s, u16* __restrict__ d, long long n) {
  long long i = (long long)blockIdx.x * 256 + threadIdx.x;
  if (i < n) d[i] = f2bf(s[i]);
}

__global__ void transpose_in(const float* __restrict__ desc, float* __restrict__ x,
                             u16* __restrict__ xb) {
  const long long n = (long long)BATCH * NPTS * DIM;
  long long i = (long long)blockIdx.x * 256 + threadIdx.x;
  if (i >= n) return;
  long long b = i / ((long long)NPTS * DIM);
  long long r = i - b * (long long)NPTS * DIM;
  long long pt = r / DIM, ch = r - pt * DIM;
  float v = desc[i];
  long long o = b * (long long)DIM * NPTS + ch * NPTS + pt;
  x[o] = v;
  xb[o] = f2bf(v);
}

__global__ void copy_x_to_y(const u16* __restrict__ x, u16* __restrict__ y) {
  const long long n = (long long)BATCH * DIM * NPTS;
  long long i = (long long)blockIdx.x * 256 + threadIdx.x;
  if (i >= n) return;
  long long b = i / ((long long)DIM * NPTS);
  long long r = i - b * (long long)DIM * NPTS;
  y[b * (long long)TWOD * NPTS + r] = x[i];
}

__global__ void residual_update(float* __restrict__ x, const float* __restrict__ d,
                                u16* __restrict__ xb, long long n) {
  long long i = (long long)blockIdx.x * 256 + threadIdx.x;
  if (i >= n) return;
  float v = x[i] + d[i];
  x[i] = v;
  xb[i] = f2bf(v);
}

__global__ __launch_bounds__(256) void softmax_rows(const float* __restrict__ S,
                                                    u16* __restrict__ P, int cols) {
  const long long row = blockIdx.x;
  const float* s = S + row * cols;
  u16* p = P + row * cols;
  __shared__ float red[256];
  const int t = threadIdx.x;
  float mx = -3.4e38f;
  for (int j = t; j < cols; j += 256) mx = fmaxf(mx, s[j]);
  red[t] = mx; __syncthreads();
  for (int st = 128; st > 0; st >>= 1) { if (t < st) red[t] = fmaxf(red[t], red[t + st]); __syncthreads(); }
  mx = red[0]; __syncthreads();
  float sum = 0.f;
  for (int j = t; j < cols; j += 256) sum += expf(s[j] - mx);
  red[t] = sum; __syncthreads();
  for (int st = 128; st > 0; st >>= 1) { if (t < st) red[t] += red[t + st]; __syncthreads(); }
  const float inv = 1.f / red[0];
  for (int j = t; j < cols; j += 256) p[j] = f2bf(expf(s[j] - mx) * inv);
}

__global__ __launch_bounds__(256) void instnorm_relu(const float* __restrict__ X,
                                                     u16* __restrict__ Z, int cols) {
  const long long row = blockIdx.x;
  const float* xr = X + row * cols;
  u16* zr = Z + row * cols;
  __shared__ float red[256];
  const int t = threadIdx.x;
  float s = 0.f;
  for (int j = t; j < cols; j += 256) s += xr[j];
  red[t] = s; __syncthreads();
  for (int st = 128; st > 0; st >>= 1) { if (t < st) red[t] += red[t + st]; __syncthreads(); }
  const float mean = red[0] / cols; __syncthreads();
  float s2 = 0.f;
  for (int j = t; j < cols; j += 256) { float d = xr[j] - mean; s2 += d * d; }
  red[t] = s2; __syncthreads();
  for (int st = 128; st > 0; st >>= 1) { if (t < st) red[t] += red[t + st]; __syncthreads(); }
  const float inv = rsqrtf(red[0] / cols + 1e-5f);
  for (int j = t; j < cols; j += 256) {
    float v = (xr[j] - mean) * inv;
    zr[j] = f2bf(v > 0.f ? v : 0.f);
  }
}

__global__ void build_couplings(const float* __restrict__ S, const float* __restrict__ alpha,
                                float* __restrict__ Cc) {
  const long long total = (long long)BATCH * MP1 * MP1;
  long long idx = (long long)blockIdx.x * 256 + threadIdx.x;
  if (idx >= total) return;
  long long b = idx / ((long long)MP1 * MP1);
  long long r = idx - b * (long long)MP1 * MP1;
  int i = (int)(r / MP1), j = (int)(r - (long long)i * MP1);
  Cc[idx] = (i < NPTS && j < NPTS) ? S[(b * NPTS + i) * NPTS + j] : alpha[0];
}

__global__ void zero_f32(float* __restrict__ p, long long n) {
  long long i = (long long)blockIdx.x * 256 + threadIdx.x;
  if (i < n) p[i] = 0.f;
}

__global__ __launch_bounds__(256) void sinkhorn_u(const float* __restrict__ Cc,
                                                  const float* __restrict__ V,
                                                  float* __restrict__ U) {
  const int i = blockIdx.x, b = blockIdx.y, t = threadIdx.x;
  const float* row = Cc + ((long long)b * MP1 + i) * MP1;
  const float* vb = V + (long long)b * MP1;
  __shared__ float red[256];
  float mx = -3.4e38f;
  for (int j = t; j < MP1; j += 256) mx = fmaxf(mx, row[j] + vb[j]);
  red[t] = mx; __syncthreads();
  for (int st = 128; st > 0; st >>= 1) { if (t < st) red[t] = fmaxf(red[t], red[t + st]); __syncthreads(); }
  mx = red[0]; __syncthreads();
  float sum = 0.f;
  for (int j = t; j < MP1; j += 256) sum += expf(row[j] + vb[j] - mx);
  red[t] = sum; __syncthreads();
  for (int st = 128; st > 0; st >>= 1) { if (t < st) red[t] += red[t + st]; __syncthreads(); }
  if (t == 0) {
    const float norm = -logf((float)(2 * NPTS));
    const float log_mu = (i < NPTS) ? norm : (logf((float)NPTS) + norm);
    U[(long long)b * MP1 + i] = log_mu - (mx + logf(red[0]));
  }
}

__global__ __launch_bounds__(256) void sinkhorn_v(const float* __restrict__ Cc,
                                                  const float* __restrict__ U,
                                                  float* __restrict__ V) {
  const int j = blockIdx.x, b = blockIdx.y, t = threadIdx.x;
  const float* ub = U + (long long)b * MP1;
  __shared__ float red[256];
  float mx = -3.4e38f;
  for (int i = t; i < MP1; i += 256) mx = fmaxf(mx, Cc[((long long)b * MP1 + i) * MP1 + j] + ub[i]);
  red[t] = mx; __syncthreads();
  for (int st = 128; st > 0; st >>= 1) { if (t < st) red[t] = fmaxf(red[t], red[t + st]); __syncthreads(); }
  mx = red[0]; __syncthreads();
  float sum = 0.f;
  for (int i = t; i < MP1; i += 256) sum += expf(Cc[((long long)b * MP1 + i) * MP1 + j] + ub[i] - mx);
  red[t] = sum; __syncthreads();
  for (int st = 128; st > 0; st >>= 1) { if (t < st) red[t] += red[t + st]; __syncthreads(); }
  if (t == 0) {
    const float norm = -logf((float)(2 * NPTS));
    const float log_nu = (j < NPTS) ? norm : (logf((float)NPTS) + norm);
    V[(long long)b * MP1 + j] = log_nu - (mx + logf(red[0]));
  }
}

__global__ void sinkhorn_final(const float* __restrict__ Cc, const float* __restrict__ U,
                               const float* __restrict__ V, float* __restrict__ out) {
  const long long total = (long long)BATCH * MP1 * MP1;
  long long idx = (long long)blockIdx.x * 256 + threadIdx.x;
  if (idx >= total) return;
  long long b = idx / ((long long)MP1 * MP1);
  long long r = idx - b * (long long)MP1 * MP1;
  int i = (int)(r / MP1), j = (int)(r - (long long)i * MP1);
  // - norm == + log(2N)
  out[idx] = Cc[idx] + U[b * MP1 + i] + V[b * MP1 + j] + logf((float)(2 * NPTS));
}

// ---------------------------------------------------------------------------
// Host orchestration
// ---------------------------------------------------------------------------
static inline dim3 ew_grid(long long n) { return dim3((unsigned)((n + 255) / 256)); }

extern "C" void kernel_launch(void* const* d_in, const int* in_sizes, int n_in,
                              void* d_out, int out_size, void* d_ws, size_t ws_size,
                              hipStream_t stream) {
  const float* descs0 = (const float*)d_in[0];
  const float* descs1 = (const float*)d_in[1];
  const float* Wq = (const float*)d_in[2];  const float* bq = (const float*)d_in[3];
  const float* Wk = (const float*)d_in[4];  const float* bk = (const float*)d_in[5];
  const float* Wv = (const float*)d_in[6];  const float* bv = (const float*)d_in[7];
  const float* Wm = (const float*)d_in[8];  const float* bm = (const float*)d_in[9];
  const float* W1 = (const float*)d_in[10]; const float* b1 = (const float*)d_in[11];
  const float* W2 = (const float*)d_in[12]; const float* b2 = (const float*)d_in[13];
  const float* Wf = (const float*)d_in[14]; const float* bfin = (const float*)d_in[15];
  const float* alpha = (const float*)d_in[16];
  float* out = (float*)d_out;
  (void)in_sizes; (void)n_in; (void)out_size; (void)ws_size;

  char* base = (char*)d_ws;
  size_t off = 0;
  auto alloc = [&](size_t bytes) -> void* {
    void* p = base + off;
    off = (off + bytes + 255) & ~(size_t)255;
    return p;
  };

  const size_t ACT = (size_t)BATCH * DIM * NPTS;            // 2,097,152
  const size_t ACT2 = (size_t)BATCH * TWOD * NPTS;          // 4,194,304
  const size_t SC = (size_t)BATCH * NHEADS * NPTS * NPTS;   // 33,554,432

  u16* wq_bf = (u16*)alloc((size_t)NLAYERS * DIM * DIM * 2);
  u16* wk_bf = (u16*)alloc((size_t)NLAYERS * DIM * DIM * 2);
  u16* wv_bf = (u16*)alloc((size_t)NLAYERS * DIM * DIM * 2);
  u16* wm_bf = (u16*)alloc((size_t)NLAYERS * DIM * DIM * 2);
  u16* w1_bf = (u16*)alloc((size_t)NLAYERS * TWOD * TWOD * 2);
  u16* w2_bf = (u16*)alloc((size_t)NLAYERS * DIM * TWOD * 2);
  u16* wf_bf = (u16*)alloc((size_t)DIM * DIM * 2);
  float* x0 = (float*)alloc(ACT * 4);
  float* x1 = (float*)alloc(ACT * 4);
  u16* x0_bf = (u16*)alloc(ACT * 2);
  u16* x1_bf = (u16*)alloc(ACT * 2);
  u16* q_bf = (u16*)alloc(ACT * 2);
  u16* k_bf = (u16*)alloc(ACT * 2);
  u16* v_bf = (u16*)alloc(ACT * 2);
  float* scores = (float*)alloc(SC * 4);
  u16* prob_bf = (u16*)alloc(SC * 2);
  u16* merged_bf = (u16*)alloc(ACT * 2);
  u16* y_bf = (u16*)alloc(ACT2 * 2);
  float* pre = (float*)alloc(ACT2 * 4);
  u16* z_bf = (u16*)alloc(ACT2 * 2);
  float* delta0 = (float*)alloc(ACT * 4);
  float* delta1 = (float*)alloc(ACT * 4);
  u16* m0_bf = (u16*)alloc(ACT * 2);
  u16* m1_bf = (u16*)alloc(ACT * 2);
  float* Cc = (float*)alloc((size_t)BATCH * MP1 * MP1 * 4);
  float* u_buf = (float*)alloc((size_t)BATCH * MP1 * 4);
  float* v_buf = (float*)alloc((size_t)BATCH * MP1 * 4);

  // --- cast weights to bf16 (done every launch: deterministic, no caching) ---
  {
    long long nqkv = (long long)NLAYERS * DIM * DIM;
    cast_bf16<<<ew_grid(nqkv), 256, 0, stream>>>(Wq, wq_bf, nqkv);
    cast_bf16<<<ew_grid(nqkv), 256, 0, stream>>>(Wk, wk_bf, nqkv);
    cast_bf16<<<ew_grid(nqkv), 256, 0, stream>>>(Wv, wv_bf, nqkv);
    cast_bf16<<<ew_grid(nqkv), 256, 0, stream>>>(Wm, wm_bf, nqkv);
    long long n1 = (long long)NLAYERS * TWOD * TWOD;
    cast_bf16<<<ew_grid(n1), 256, 0, stream>>>(W1, w1_bf, n1);
    long long n2 = (long long)NLAYERS * DIM * TWOD;
    cast_bf16<<<ew_grid(n2), 256, 0, stream>>>(W2, w2_bf, n2);
    long long nf = (long long)DIM * DIM;
    cast_bf16<<<ew_grid(nf), 256, 0, stream>>>(Wf, wf_bf, nf);
  }

  // --- transpose inputs to channels-first + bf16 ---
  {
    long long nx = (long long)BATCH * NPTS * DIM;
    transpose_in<<<ew_grid(nx), 256, 0, stream>>>(descs0, x0, x0_bf);
    transpose_in<<<ew_grid(nx), 256, 0, stream>>>(descs1, x1, x1_bf);
  }

  const long long sAct = (long long)DIM * NPTS;
  const long long sAct2 = (long long)TWOD * NPTS;
  const long long sHead = (long long)HDIM * NPTS;
  const long long sScore = (long long)NPTS * NPTS;

  auto prop = [&](const u16* xbf, const u16* srcbf, float* delta, int li) {
    const u16* wq_l = wq_bf + (size_t)li * DIM * DIM;
    const u16* wk_l = wk_bf + (size_t)li * DIM * DIM;
    const u16* wv_l = wv_bf + (size_t)li * DIM * DIM;
    const u16* wm_l = wm_bf + (size_t)li * DIM * DIM;
    const u16* w1_l = w1_bf + (size_t)li * TWOD * TWOD;
    const u16* w2_l = w2_bf + (size_t)li * DIM * TWOD;
    const float* bq_l = bq + (size_t)li * DIM;
    const float* bk_l = bk + (size_t)li * DIM;
    const float* bv_l = bv + (size_t)li * DIM;
    const float* bm_l = bm + (size_t)li * DIM;
    const float* b1_l = b1 + (size_t)li * TWOD;
    const float* b2_l = b2 + (size_t)li * DIM;

    // Q = Wq x ; K = Wk src ; V = Wv src  (bf16 outputs, [b][512][1024])
    gemm_wmma<false, false, true, false, true><<<dim3(8, 4, BATCH), 256, 0, stream>>>(
        wq_l, 0, DIM, xbf, sAct, NPTS, nullptr, q_bf, 1, sAct, 0, NPTS, bq_l, 1.f, DIM, NPTS, DIM);
    gemm_wmma<false, false, true, false, true><<<dim3(8, 4, BATCH), 256, 0, stream>>>(
        wk_l, 0, DIM, srcbf, sAct, NPTS, nullptr, k_bf, 1, sAct, 0, NPTS, bk_l, 1.f, DIM, NPTS, DIM);
    gemm_wmma<false, false, true, false, true><<<dim3(8, 4, BATCH), 256, 0, stream>>>(
        wv_l, 0, DIM, srcbf, sAct, NPTS, nullptr, v_bf, 1, sAct, 0, NPTS, bv_l, 1.f, DIM, NPTS, DIM);

    // scores[z=b*h] = (Q_h^T K_h)/8 : A transposed (A[k*lda+m] = q[d][n])
    gemm_wmma<true, false, false, true, false><<<dim3(8, 8, BATCH * NHEADS), 256, 0, stream>>>(
        q_bf, sHead, NPTS, k_bf, sHead, NPTS, scores, nullptr,
        1, sScore, 0, NPTS, nullptr, 0.125f, NPTS, NPTS, HDIM);

    softmax_rows<<<dim3(BATCH * NHEADS * NPTS), 256, 0, stream>>>(scores, prob_bf, NPTS);

    // O_h = V_h P^T with fused head-merge: channel = d*H + h (ldc = H*N)
    gemm_wmma<false, true, false, false, true><<<dim3(8, 1, BATCH * NHEADS), 256, 0, stream>>>(
        v_bf, sHead, NPTS, prob_bf, sScore, NPTS, nullptr, merged_bf,
        NHEADS, sAct, (long long)NPTS, NHEADS * NPTS, nullptr, 1.f, HDIM, NPTS, NPTS);

    // y = [x ; Wm merged + bm]  (bf16, [b][1024][1024])
    copy_x_to_y<<<ew_grid((long long)ACT), 256, 0, stream>>>(xbf, y_bf);
    gemm_wmma<false, false, true, false, true><<<dim3(8, 4, BATCH), 256, 0, stream>>>(
        wm_l, 0, DIM, merged_bf, sAct, NPTS, nullptr, y_bf + (size_t)DIM * NPTS,
        1, sAct2, 0, NPTS, bm_l, 1.f, DIM, NPTS, DIM);

    // pre = W1 y + b1  (f32 for InstanceNorm statistics)
    gemm_wmma<false, false, true, true, false><<<dim3(8, 8, BATCH), 256, 0, stream>>>(
        w1_l, 0, TWOD, y_bf, sAct2, NPTS, pre, nullptr,
        1, sAct2, 0, NPTS, b1_l, 1.f, TWOD, NPTS, TWOD);

    instnorm_relu<<<dim3(BATCH * TWOD), 256, 0, stream>>>(pre, z_bf, NPTS);

    // delta = W2 z + b2 (f32)
    gemm_wmma<false, false, true, true, false><<<dim3(8, 4, BATCH), 256, 0, stream>>>(
        w2_l, 0, TWOD, z_bf, sAct2, NPTS, delta, nullptr,
        1, sAct, 0, NPTS, b2_l, 1.f, DIM, NPTS, TWOD);
  };

  for (int li = 0; li < NLAYERS; ++li) {
    const bool cross = (li & 1) != 0;  // NAMES = ['self','cross']*6
    const u16* s0 = cross ? x1_bf : x0_bf;
    const u16* s1 = cross ? x0_bf : x1_bf;
    prop(x0_bf, s0, delta0, li);
    prop(x1_bf, s1, delta1, li);
    residual_update<<<ew_grid((long long)ACT), 256, 0, stream>>>(x0, delta0, x0_bf, (long long)ACT);
    residual_update<<<ew_grid((long long)ACT), 256, 0, stream>>>(x1, delta1, x1_bf, (long long)ACT);
  }

  // final projections m0/m1 = Wf x + bf (bf16)
  gemm_wmma<false, false, true, false, true><<<dim3(8, 4, BATCH), 256, 0, stream>>>(
      wf_bf, 0, DIM, x0_bf, sAct, NPTS, nullptr, m0_bf, 1, sAct, 0, NPTS, bfin, 1.f, DIM, NPTS, DIM);
  gemm_wmma<false, false, true, false, true><<<dim3(8, 4, BATCH), 256, 0, stream>>>(
      wf_bf, 0, DIM, x1_bf, sAct, NPTS, nullptr, m1_bf, 1, sAct, 0, NPTS, bfin, 1.f, DIM, NPTS, DIM);

  // scores[b,n,m] = m0^T m1 / sqrt(D)  (reuse big scores buffer, first 4 slabs)
  gemm_wmma<true, false, false, true, false><<<dim3(8, 8, BATCH), 256, 0, stream>>>(
      m0_bf, sAct, NPTS, m1_bf, sAct, NPTS, scores, nullptr,
      1, sScore, 0, NPTS, nullptr, 0.04419417382415922f, NPTS, NPTS, DIM);

  // Sinkhorn in log space
  const long long ctot = (long long)BATCH * MP1 * MP1;
  build_couplings<<<ew_grid(ctot), 256, 0, stream>>>(scores, alpha, Cc);
  zero_f32<<<ew_grid((long long)BATCH * MP1), 256, 0, stream>>>(u_buf, (long long)BATCH * MP1);
  zero_f32<<<ew_grid((long long)BATCH * MP1), 256, 0, stream>>>(v_buf, (long long)BATCH * MP1);
  for (int it = 0; it < 100; ++it) {
    sinkhorn_u<<<dim3(MP1, BATCH), 256, 0, stream>>>(Cc, v_buf, u_buf);
    sinkhorn_v<<<dim3(MP1, BATCH), 256, 0, stream>>>(Cc, u_buf, v_buf);
  }
  sinkhorn_final<<<ew_grid(ctot), 256, 0, stream>>>(Cc, u_buf, v_buf, out);
}
